// DeGCN_81243601371936
// MI455X (gfx1250) — compile-verified, matrix operand order checked by reading, not run on hardware
//
#include <hip/hip_runtime.h>
#include <hip/hip_bf16.h>

#define N_NODES  100000
#define N_EDGES  1600000
#define NFEAT    256
#define NHID     128
#define NCLASS   40
#define NCPAD    48
#define N_BRANCH 3

typedef __attribute__((ext_vector_type(16))) __bf16 v16bf;
typedef __attribute__((ext_vector_type(8)))  __bf16 v8bf;
typedef __attribute__((ext_vector_type(8)))  float  v8f;

// ---------------------------------------------------------------------------
// float32 -> bf16 conversion (for x)
// ---------------------------------------------------------------------------
__global__ void cvt_f32_bf16_kernel(const float* __restrict__ src,
                                    __bf16* __restrict__ dst, long long n) {
    long long i = (long long)blockIdx.x * blockDim.x + threadIdx.x;
    if (i < n) dst[i] = (__bf16)src[i];
}

// ---------------------------------------------------------------------------
// Repack W1 [3,256,128] f32 into WMMA B-fragment-major bf16:
//   w1f[(((br*8 + ct)*8 + ks)*32 + lane)*16 + e]
//     = W1[br][ks*32 + 16*(e/8) + 8*(lane/16) + e%8][ct*16 + lane%16]
// so the GEMM inner loop loads each lane's 16 B elements contiguously.
// ---------------------------------------------------------------------------
__global__ void repack_w1_kernel(const float* __restrict__ W1,
                                 __bf16* __restrict__ w1f) {
    int i = blockIdx.x * blockDim.x + threadIdx.x;
    if (i >= N_BRANCH * 8 * 8 * 32 * 16) return;
    const int e    = i & 15;
    const int lane = (i >> 4) & 31;
    const int ks   = (i >> 9) & 7;
    const int ct   = (i >> 12) & 7;
    const int br   = i >> 15;
    const int hv = lane >> 4, l16 = lane & 15;
    const int h    = ct * 16 + l16;
    const int kidx = ks * 32 + 16 * (e >> 3) + 8 * hv + (e & 7);
    w1f[i] = (__bf16)W1[((size_t)br * NFEAT + kidx) * NHID + h];
}

// Repack W2 [128,40] f32 into fragment-major bf16, padded to 48 cols:
//   w2f[((ct*4 + ks)*32 + lane)*16 + e]
__global__ void repack_w2_kernel(const float* __restrict__ W2,
                                 __bf16* __restrict__ w2f) {
    int i = blockIdx.x * blockDim.x + threadIdx.x;
    if (i >= 3 * 4 * 32 * 16) return;
    const int e    = i & 15;
    const int lane = (i >> 4) & 31;
    const int ks   = (i >> 9) & 3;
    const int ct   = i >> 11;
    const int hv = lane >> 4, l16 = lane & 15;
    const int h    = ct * 16 + l16;
    const int kidx = ks * 32 + 16 * (e >> 3) + 8 * hv + (e & 7);
    w2f[i] = (h < NCLASS) ? (__bf16)W2[kidx * NCLASS + h] : (__bf16)0.0f;
}

// hsum[n][j] = b1[0][j] + b1[1][j] + b1[2][j]  (bias folded into accumulator)
__global__ void init_hsum_kernel(const float* __restrict__ b1,
                                 float* __restrict__ hsum) {
    long long i = (long long)blockIdx.x * blockDim.x + threadIdx.x;
    if (i >= (long long)N_NODES * NHID) return;
    int j = (int)(i % NHID);
    hsum[i] = b1[j] + b1[NHID + j] + b1[2 * NHID + j];
}

// logits[n][c] = b2[c]
__global__ void init_logits_kernel(const float* __restrict__ b2,
                                   float* __restrict__ logits) {
    long long i = (long long)blockIdx.x * blockDim.x + threadIdx.x;
    if (i >= (long long)N_NODES * NCLASS) return;
    logits[i] = b2[(int)(i % NCLASS)];
}

// relu + bf16 convert: hb = bf16(max(hsum, 0))
__global__ void relu_cvt_kernel(const float* __restrict__ hsum,
                                __bf16* __restrict__ hb) {
    long long i = (long long)blockIdx.x * blockDim.x + threadIdx.x;
    if (i < (long long)N_NODES * NHID) hb[i] = (__bf16)fmaxf(hsum[i], 0.0f);
}

// ---------------------------------------------------------------------------
// GEMM1: support[br] = x @ W1[br]   (100000x256 @ 256x128, 3 branches)
// Block = 2 waves; each wave computes a 16x64 strip (4 accumulators share one
// A fragment -> 4 WMMAs per A load). grid = (N_NODES/16 = 6250, 3).
// A layout (16-bit 16x32): lane = half*16 + m; element i holds
//   K = 16*(i/8) + 8*half + (i%8)  -> two contiguous 16B loads per lane.
// ---------------------------------------------------------------------------
__global__ void __launch_bounds__(64) gemm1_wmma_kernel(
        const __bf16* __restrict__ xb, const __bf16* __restrict__ w1f,
        float* __restrict__ support) {
    const int wave = threadIdx.x >> 5;   // 0..1 -> which half of NHID
    const int lane = threadIdx.x & 31;
    const int hv   = lane >> 4;
    const int l16  = lane & 15;
    const int rowTile = blockIdx.x * 16;
    const int br      = blockIdx.y;

    const __bf16* __restrict__ xrow  = xb + (size_t)(rowTile + l16) * NFEAT;
    // base of this wave's 4 col-tile fragment blocks (each block: 8*32*16 elems)
    const __bf16* __restrict__ fbase =
        w1f + ((size_t)br * 8 + wave * 4) * (8 * 32 * 16);
    const int laneoff = lane * 16;

    v8f acc[4] = {};
#pragma unroll
    for (int ks = 0; ks < 8; ++ks) {
        v8bf lo = *(const v8bf*)(xrow + ks * 32 + 8 * hv);
        v8bf hi = *(const v8bf*)(xrow + ks * 32 + 16 + 8 * hv);
        v16bf a;
#pragma unroll
        for (int i = 0; i < 8; ++i) { a[i] = lo[i]; a[8 + i] = hi[i]; }
#pragma unroll
        for (int t = 0; t < 4; ++t) {
            v16bf b = *(const v16bf*)(fbase + (t * 8 + ks) * 512 + laneoff);
            acc[t] = __builtin_amdgcn_wmma_f32_16x16x32_bf16(
                         false, a, false, b, (short)0, acc[t], false, false);
        }
    }
    float* __restrict__ out = support + (size_t)br * N_NODES * NHID;
#pragma unroll
    for (int t = 0; t < 4; ++t) {
        const int col = (wave * 4 + t) * 16 + l16;
#pragma unroll
        for (int r = 0; r < 8; ++r) {
            out[(size_t)(rowTile + hv * 8 + r) * NHID + col] = acc[t][r];
        }
    }
}

// ---------------------------------------------------------------------------
// GEMM2: hw2 = relu(h) @ W2pad   (100000x128 @ 128x48)
// One wave per block computes the full 16x48 strip (3 accumulators).
// grid = (6250)
// ---------------------------------------------------------------------------
__global__ void __launch_bounds__(32) gemm2_wmma_kernel(
        const __bf16* __restrict__ hb, const __bf16* __restrict__ w2f,
        float* __restrict__ hw2) {
    const int lane = threadIdx.x & 31;
    const int hv   = lane >> 4;
    const int l16  = lane & 15;
    const int rowTile = blockIdx.x * 16;

    const __bf16* __restrict__ hrow = hb + (size_t)(rowTile + l16) * NHID;
    const int laneoff = lane * 16;

    v8f acc[3] = {};
#pragma unroll
    for (int ks = 0; ks < 4; ++ks) {
        v8bf lo = *(const v8bf*)(hrow + ks * 32 + 8 * hv);
        v8bf hi = *(const v8bf*)(hrow + ks * 32 + 16 + 8 * hv);
        v16bf a;
#pragma unroll
        for (int i = 0; i < 8; ++i) { a[i] = lo[i]; a[8 + i] = hi[i]; }
#pragma unroll
        for (int ct = 0; ct < 3; ++ct) {
            v16bf b = *(const v16bf*)(w2f + (ct * 4 + ks) * 512 + laneoff);
            acc[ct] = __builtin_amdgcn_wmma_f32_16x16x32_bf16(
                          false, a, false, b, (short)0, acc[ct], false, false);
        }
    }
#pragma unroll
    for (int ct = 0; ct < 3; ++ct) {
#pragma unroll
        for (int r = 0; r < 8; ++r) {
            hw2[(size_t)(rowTile + hv * 8 + r) * NCPAD + ct * 16 + l16] =
                acc[ct][r];
        }
    }
}

// ---------------------------------------------------------------------------
// SpMM1: hsum[row] += val * support[br][col]   (32 thr/edge, 4 feats each ->
// float4 gather + 4 f32 atomics). grid = (200000, 3); branch = blockIdx.y.
// ---------------------------------------------------------------------------
__global__ void __launch_bounds__(256) spmm1_kernel(
        const int* __restrict__ sub_row, const int* __restrict__ sub_col,
        const float* __restrict__ sub_val, const float* __restrict__ support,
        float* __restrict__ hsum) {
    const unsigned gid  = blockIdx.x * 256u + threadIdx.x;
    const int      part = (int)(gid & 31);
    const unsigned e    = gid >> 5;                // edge within branch
    const int      br   = blockIdx.y;
    const size_t   ke   = (size_t)br * N_EDGES + e;

    // streaming edge lists: prefetch ahead (global_prefetch_b8)
    __builtin_prefetch(sub_col + ke + 16384, 0, 1);
    __builtin_prefetch(sub_row + ke + 16384, 0, 1);

    const int   r  = sub_row[ke];
    const int   cn = sub_col[ke];
    const float v  = sub_val[ke];
    const float4 s = *(const float4*)(support +
        ((size_t)br * N_NODES + cn) * NHID + part * 4);
    float* dst = hsum + (size_t)r * NHID + part * 4;
    atomicAdd(dst + 0, v * s.x);
    atomicAdd(dst + 1, v * s.y);
    atomicAdd(dst + 2, v * s.z);
    atomicAdd(dst + 3, v * s.w);
}

// ---------------------------------------------------------------------------
// SpMM2: logits[row] += val * hw2[col]   (8 thr/edge, 5 features each)
// grid exact: N_EDGES*8/256 = 50000 blocks
// ---------------------------------------------------------------------------
__global__ void __launch_bounds__(256) spmm2_kernel(
        const int* __restrict__ adj_row, const int* __restrict__ adj_col,
        const float* __restrict__ adj_val, const float* __restrict__ hw2,
        float* __restrict__ logits) {
    const unsigned gid = blockIdx.x * 256u + threadIdx.x;
    const int      t   = (int)(gid & 7);
    const unsigned e   = gid >> 3;

    __builtin_prefetch(adj_col + e + 16384, 0, 1);

    const int   r  = adj_row[e];
    const int   cn = adj_col[e];
    const float v  = adj_val[e];
    const float* src = hw2 + (size_t)cn * NCPAD + t * 5;
    float*       dst = logits + (size_t)r * NCLASS + t * 5;
#pragma unroll
    for (int i = 0; i < 5; ++i) atomicAdd(dst + i, v * src[i]);
}

// ---------------------------------------------------------------------------
// Rowwise log-softmax over 40 classes; one wave32 per row (8 rows / block)
// ---------------------------------------------------------------------------
__global__ void __launch_bounds__(256) log_softmax_kernel(
        const float* __restrict__ logits, float* __restrict__ out) {
    const int row  = blockIdx.x * 8 + (threadIdx.x >> 5);
    const int lane = threadIdx.x & 31;
    if (row >= N_NODES) return;
    const float* p = logits + (size_t)row * NCLASS;
    const float x0 = p[lane];
    const float x1 = (lane + 32 < NCLASS) ? p[lane + 32] : -INFINITY;
    float m = fmaxf(x0, x1);
#pragma unroll
    for (int off = 16; off > 0; off >>= 1) m = fmaxf(m, __shfl_xor(m, off, 32));
    float s = __expf(x0 - m) + ((lane + 32 < NCLASS) ? __expf(x1 - m) : 0.0f);
#pragma unroll
    for (int off = 16; off > 0; off >>= 1) s += __shfl_xor(s, off, 32);
    const float lse = m + __logf(s);
    float* o = out + (size_t)row * NCLASS;
    o[lane] = x0 - lse;
    if (lane + 32 < NCLASS) o[lane + 32] = x1 - lse;
}

// ---------------------------------------------------------------------------
extern "C" void kernel_launch(void* const* d_in, const int* in_sizes, int n_in,
                              void* d_out, int out_size, void* d_ws, size_t ws_size,
                              hipStream_t stream) {
    const float* x       = (const float*)d_in[0];
    const int*   adj_row = (const int*)  d_in[1];
    const int*   adj_col = (const int*)  d_in[2];
    const float* adj_val = (const float*)d_in[3];
    const int*   sub_row = (const int*)  d_in[4];
    const int*   sub_col = (const int*)  d_in[5];
    const float* sub_val = (const float*)d_in[6];
    const float* W1      = (const float*)d_in[7];
    const float* b1      = (const float*)d_in[8];
    const float* W2      = (const float*)d_in[9];
    const float* b2      = (const float*)d_in[10];
    float* out = (float*)d_out;

    // workspace carve-up (256B aligned)
    char*  base = (char*)d_ws;
    size_t off  = 0;
    auto carve = [&](size_t bytes) {
        size_t o = off;
        off = (off + bytes + 255) & ~(size_t)255;
        return o;
    };
    __bf16* xb      = (__bf16*)(base + carve((size_t)N_NODES * NFEAT * 2));
    __bf16* w1f     = (__bf16*)(base + carve((size_t)N_BRANCH * 8 * 8 * 32 * 16 * 2));
    __bf16* w2f     = (__bf16*)(base + carve((size_t)3 * 4 * 32 * 16 * 2));
    float*  support = (float*) (base + carve((size_t)N_BRANCH * N_NODES * NHID * 4));
    float*  hsum    = (float*) (base + carve((size_t)N_NODES * NHID * 4));
    __bf16* hb      = (__bf16*)(base + carve((size_t)N_NODES * NHID * 2));
    float*  hw2     = (float*) (base + carve((size_t)N_NODES * NCPAD * 4));
    float*  logits  = (float*) (base + carve((size_t)N_NODES * NCLASS * 4));
    (void)ws_size; (void)n_in; (void)in_sizes; (void)out_size;

    const int B = 256;
    // precision conversion of x + weight fragment repacks
    {
        long long n = (long long)N_NODES * NFEAT;
        cvt_f32_bf16_kernel<<<(int)((n + B - 1) / B), B, 0, stream>>>(x, xb, n);
    }
    repack_w1_kernel<<<(N_BRANCH * 8 * 8 * 32 * 16 + B - 1) / B, B, 0, stream>>>(W1, w1f);
    repack_w2_kernel<<<(3 * 4 * 32 * 16 + B - 1) / B, B, 0, stream>>>(W2, w2f);

    // GEMM1 (WMMA): support[br] = x @ W1[br]
    {
        dim3 grid(N_NODES / 16, N_BRANCH);
        gemm1_wmma_kernel<<<grid, 64, 0, stream>>>(xb, w1f, support);
    }

    // hsum = sum_k b1[k]  (bias init), then scatter the 3 branch SpMMs into it
    {
        long long n = (long long)N_NODES * NHID;
        init_hsum_kernel<<<(int)((n + B - 1) / B), B, 0, stream>>>(b1, hsum);
    }
    {
        dim3 grid(N_EDGES * 32 / 256, N_BRANCH);   // 200000 x 3, exact
        spmm1_kernel<<<grid, 256, 0, stream>>>(sub_row, sub_col, sub_val,
                                               support, hsum);
    }

    // relu + bf16 convert
    {
        long long n = (long long)N_NODES * NHID;
        relu_cvt_kernel<<<(int)((n + B - 1) / B), B, 0, stream>>>(hsum, hb);
    }

    // GEMM2 (WMMA): hw2 = h @ W2pad
    gemm2_wmma_kernel<<<N_NODES / 16, 32, 0, stream>>>(hb, w2f, hw2);

    // logits = b2, then scatter SpMM2 into it
    {
        long long n = (long long)N_NODES * NCLASS;
        init_logits_kernel<<<(int)((n + B - 1) / B), B, 0, stream>>>(b2, logits);
    }
    spmm2_kernel<<<N_EDGES * 8 / 256, 256, 0, stream>>>(adj_row, adj_col,
                                                        adj_val, hw2, logits);

    // log-softmax into d_out
    log_softmax_kernel<<<(N_NODES + 7) / 8, 256, 0, stream>>>(logits, out);
}
